// MaLstmCell_39092792328850
// MI455X (gfx1250) — compile-verified, matrix operand order checked by reading, not run on hardware
//
#include <hip/hip_runtime.h>
#include <math.h>

#define B_ 64
#define T_ 512
#define D_ 1024
#define H_ 1024

#define KC 64          // K elements staged per chunk (per stream)
#define NCHUNK 16      // H_ / KC  (both streams staged in parallel)

typedef __bf16 bf16;
typedef __attribute__((ext_vector_type(16))) __bf16 v16bf;
typedef __attribute__((ext_vector_type(8)))  __bf16 v8bf;
typedef __attribute__((ext_vector_type(8)))  float  v8f;

// ---------------- helpers ----------------

__device__ __forceinline__ unsigned short f32_to_bf16(float f) {
  union { float f; unsigned u; } a; a.f = f;
  unsigned r = a.u + 0x7FFFu + ((a.u >> 16) & 1u);   // round-to-nearest-even
  return (unsigned short)(r >> 16);
}

__device__ __forceinline__ float sigm(float x) {
  return 1.0f / (1.0f + __expf(-x));
}

__device__ __forceinline__ float tanh_fast(float x) {
  return 2.0f * sigm(2.0f * x) - 1.0f;   // hardware v_exp_f32 path
}

// Generic->LDS byte offset (LDS aperture uses addr[31:0]).
__device__ __forceinline__ unsigned lds_off(const void* p) {
  return (unsigned)(unsigned long long)(uintptr_t)p;
}

// Build a 16x32 bf16 WMMA fragment from LDS.
// Per-lane: two contiguous 8 x bf16 (16 B) runs at +base and +base+16
// (ISA 7.12.2 layout: lanes 0-15 K-base 0, lanes 16-31 K-base 8).
__device__ __forceinline__ v16bf lds_frag(const unsigned short* p) {
  const bf16* q = (const bf16*)p;
  v8bf lo = *(const v8bf*)(q);
  v8bf hi = *(const v8bf*)(q + 16);
  v16bf r;
#pragma unroll
  for (int j = 0; j < 8; ++j) { r[j] = lo[j]; r[j + 8] = hi[j]; }
  return r;
}

__device__ __forceinline__ v8f wmma_bf16(v16bf a, v16bf b, v8f c) {
  return __builtin_amdgcn_wmma_f32_16x16x32_bf16(false, a, false, b,
                                                 (short)0, c, false, false);
}

// ---------------- small kernels ----------------

__global__ void cvt_f32_bf16(const float* __restrict__ src,
                             unsigned short* __restrict__ dst, int n) {
  int i = blockIdx.x * blockDim.x + threadIdx.x;
  if (i < n) dst[i] = f32_to_bf16(src[i]);
}

__global__ void init_state(unsigned short* __restrict__ hA,
                           unsigned short* __restrict__ hB,
                           float* __restrict__ h32,
                           float* __restrict__ c32, int n) {
  int i = blockIdx.x * blockDim.x + threadIdx.x;
  if (i < n) { hA[i] = 0; hB[i] = 0; h32[i] = 0.0f; c32[i] = 0.0f; }
}

__global__ void copy_finals(const float* __restrict__ h32,
                            const float* __restrict__ c32,
                            float* __restrict__ dst, int n) {
  int i = blockIdx.x * blockDim.x + threadIdx.x;
  if (i < n) { dst[i] = h32[i]; dst[n + i] = c32[i]; }
}

// ---------------- LSTM timestep ----------------
// 64 blocks x 256 threads (8 waves, 2 per SIMD). Block -> one 16-wide hidden
// tile. Wave-group 0 (waves 0-3): recurrent stream h . Ws^T (K=1024).
// Wave-group 1 (waves 4-7): input stream x_t . Wi^T (K=1024).
// Within a group, wave w -> batch tile b0 = 16*(w&3). Both streams are staged
// per KC=64 chunk into double-buffered LDS via global_load_async_to_lds_b128;
// group-1 partials are reduced into group-0 through LDS, group 0 runs the
// f32 gate/state epilogue.
//
// LDS per buffer (32 KB):
//   A_h [64][KC] | W_s [4][16][KC] | A_x [64][KC] | W_i [4][16][KC]
__global__ __launch_bounds__(256) void lstm_step(
    const unsigned short* __restrict__ x_bfp,
    const unsigned short* __restrict__ Wi_bfp,
    const unsigned short* __restrict__ Ws_bfp,
    const float* __restrict__ bias,      // [4, H]
    const int* __restrict__ lengths,     // [B]
    const unsigned short* __restrict__ h_rdp,
    unsigned short* __restrict__ h_wrp,
    float* __restrict__ h32,
    float* __restrict__ c32,
    float* __restrict__ out,             // [B, T, H]
    int t) {
  __shared__ __align__(16) unsigned short sbuf[2][16384];  // 2 x 32 KB

  const int tid   = threadIdx.x;
  const int lane  = tid & 31;
  const int wave  = tid >> 5;          // 0..7
  const int group = wave >> 2;         // 0: h/Ws stream, 1: x/Wi stream
  const int w4    = wave & 3;          // batch tile within group
  const int col0  = blockIdx.x << 4;   // hidden tile origin
  const int b0    = w4 << 4;

  v8f acc0 = {0.f,0.f,0.f,0.f,0.f,0.f,0.f,0.f};
  v8f acc1 = acc0, acc2 = acc0, acc3 = acc0;

  // ---- precompute per-thread staging addresses (8 x 16B transfers/chunk) ----
  // j -> region is compile-time: reg = j>>1 (0:A_h 1:W_s 2:A_x 3:W_i),
  // idx = (j&1)*256 + tid in [0,512). No divergent control flow.
  unsigned loff0[8];
  const unsigned short* gbase[8];
#pragma unroll
  for (int j = 0; j < 8; ++j) {
    const int reg = j >> 1;
    const int idx = ((j & 1) << 8) + tid;
    if (reg == 0 || reg == 2) {            // A regions: 64 rows x 8 x 16B
      const int row = idx >> 3;
      const int ke  = (idx & 7) << 3;
      gbase[j] = (reg == 0)
          ? (h_rdp + (size_t)row * H_ + ke)
          : (x_bfp + ((size_t)row * T_ + (size_t)t) * D_ + ke);
      loff0[j] = lds_off(&sbuf[0][(reg << 12) + row * KC + ke]);
    } else {                               // W regions: 4g x 16r x 8 x 16B
      const int g  = idx >> 7;
      const int r  = (idx >> 3) & 15;
      const int ke = (idx & 7) << 3;
      const unsigned short* Wsrc = (reg == 1) ? Ws_bfp : Wi_bfp;
      gbase[j] = Wsrc + (size_t)(g * H_ + col0 + r) * 1024 + ke;
      loff0[j] = lds_off(&sbuf[0][(reg << 12) + (g * 16 + r) * KC + ke]);
    }
  }

  auto issue = [&](int c, int p) {
    const unsigned long long kbyte = (unsigned long long)(c * KC * 2);
    const unsigned lbuf = (unsigned)p * (16384u * 2u);
#pragma unroll
    for (int j = 0; j < 8; ++j) {
      asm volatile("global_load_async_to_lds_b128 %0, %1, off"
                   :: "v"(loff0[j] + lbuf),
                      "v"((unsigned long long)(uintptr_t)gbase[j] + kbyte)
                   : "memory");
    }
  };

  // ---- consume one staged chunk: batch fragment loads, 8 WMMAs ----
  const int Aoff = group ? 8192 : 0;
  const int Woff = group ? 12288 : 4096;
  const int m    = lane & 15;
  const int base = ((lane >> 4) & 1) << 3;
  auto consume = [&](int p) {
    const unsigned short* A = &sbuf[p][Aoff] + (b0 + m) * KC + base;
    const unsigned short* W = &sbuf[p][Woff] + m * KC + base;
    v16bf a0  = lds_frag(A);
    v16bf a1  = lds_frag(A + 32);
    v16bf b00 = lds_frag(W + 0 * 16 * KC);
    v16bf b10 = lds_frag(W + 1 * 16 * KC);
    v16bf b20 = lds_frag(W + 2 * 16 * KC);
    v16bf b30 = lds_frag(W + 3 * 16 * KC);
    v16bf b01 = lds_frag(W + 0 * 16 * KC + 32);
    v16bf b11 = lds_frag(W + 1 * 16 * KC + 32);
    v16bf b21 = lds_frag(W + 2 * 16 * KC + 32);
    v16bf b31 = lds_frag(W + 3 * 16 * KC + 32);
    acc0 = wmma_bf16(a0, b00, acc0);
    acc1 = wmma_bf16(a0, b10, acc1);
    acc2 = wmma_bf16(a0, b20, acc2);
    acc3 = wmma_bf16(a0, b30, acc3);
    acc0 = wmma_bf16(a1, b01, acc0);
    acc1 = wmma_bf16(a1, b11, acc1);
    acc2 = wmma_bf16(a1, b21, acc2);
    acc3 = wmma_bf16(a1, b31, acc3);
  };

  // ---- software-pipelined main loop ----
  issue(0, 0);
  asm volatile("s_wait_asynccnt 0" ::: "memory");
  __syncthreads();
#pragma unroll 2
  for (int c = 0; c < NCHUNK; ++c) {
    const int p = c & 1;
    if (c + 1 < NCHUNK) issue(c + 1, p ^ 1);
    consume(p);
    asm volatile("s_wait_asynccnt 0" ::: "memory");  // my writes to p^1 done
    __syncthreads();            // all waves done reading p / writing p^1
  }

  // ---- cross-group reduction through LDS (staging buffers now free) ----
  v8f* red = (v8f*)&sbuf[0][0];   // 4 waves x 4 gates x 32 lanes x v8f = 16 KB
  if (group == 1) {
    red[((w4 * 4 + 0) << 5) + lane] = acc0;
    red[((w4 * 4 + 1) << 5) + lane] = acc1;
    red[((w4 * 4 + 2) << 5) + lane] = acc2;
    red[((w4 * 4 + 3) << 5) + lane] = acc3;
  }
  __syncthreads();
  if (group == 1) return;

  acc0 = acc0 + red[((w4 * 4 + 0) << 5) + lane];
  acc1 = acc1 + red[((w4 * 4 + 1) << 5) + lane];
  acc2 = acc2 + red[((w4 * 4 + 2) << 5) + lane];
  acc3 = acc3 + red[((w4 * 4 + 3) << 5) + lane];

  // ---- epilogue: gates, state update, masking, stores (f32) ----
  // C/D layout: lane -> N = lane&15; element e -> M = (lane>=16 ? 8 : 0)+e.
  const int hcol  = col0 + (lane & 15);
  const int rbase = b0 + (((lane >> 4) & 1) << 3);
  const float bi = bias[0 * H_ + hcol];
  const float bf = bias[1 * H_ + hcol];
  const float bm = bias[2 * H_ + hcol];
  const float bo = bias[3 * H_ + hcol];
#pragma unroll
  for (int e = 0; e < 8; ++e) {
    const int brow = rbase + e;
    const size_t idx = (size_t)brow * H_ + hcol;
    const float iv = sigm(acc0[e] + bi);
    const float fv = sigm(acc1[e] + bf);
    const float mv = tanh_fast(acc2[e] + bm);
    const float ov = sigm(acc3[e] + bo);
    const float cold = c32[idx];
    const float cn = iv * mv + fv * cold;
    const float hn = ov * tanh_fast(cn);
    const bool valid = t < lengths[brow];
    const float cs = valid ? cn : cold;
    const float hs = valid ? hn : h32[idx];
    c32[idx] = cs;
    h32[idx] = hs;
    h_wrp[idx] = f32_to_bf16(hs);
    out[((size_t)brow * T_ + (size_t)t) * H_ + hcol] = valid ? hn : 0.0f;
  }
}

// ---------------- host ----------------

extern "C" void kernel_launch(void* const* d_in, const int* in_sizes, int n_in,
                              void* d_out, int out_size, void* d_ws, size_t ws_size,
                              hipStream_t stream) {
  const float* x    = (const float*)d_in[0];
  const int* lens   = (const int*)d_in[1];
  const float* Wi   = (const float*)d_in[2];
  const float* Ws   = (const float*)d_in[3];
  const float* bias = (const float*)d_in[4];
  float* out        = (float*)d_out;

  char* w = (char*)d_ws;
  unsigned short* x_bf  = (unsigned short*)w; w += (size_t)B_ * T_ * D_ * 2;  // 64 MB
  unsigned short* Wi_bf = (unsigned short*)w; w += (size_t)4 * H_ * D_ * 2;   //  8 MB
  unsigned short* Ws_bf = (unsigned short*)w; w += (size_t)4 * H_ * H_ * 2;   //  8 MB
  unsigned short* hA    = (unsigned short*)w; w += (size_t)B_ * H_ * 2;
  unsigned short* hB    = (unsigned short*)w; w += (size_t)B_ * H_ * 2;
  float* h32            = (float*)w;          w += (size_t)B_ * H_ * 4;
  float* c32            = (float*)w;          w += (size_t)B_ * H_ * 4;
  (void)in_sizes; (void)n_in; (void)out_size; (void)ws_size;

  {
    int n = B_ * T_ * D_;
    cvt_f32_bf16<<<(n + 255) / 256, 256, 0, stream>>>(x, x_bf, n);
  }
  {
    int n = 4 * H_ * D_;
    cvt_f32_bf16<<<(n + 255) / 256, 256, 0, stream>>>(Wi, Wi_bf, n);
  }
  {
    int n = 4 * H_ * H_;
    cvt_f32_bf16<<<(n + 255) / 256, 256, 0, stream>>>(Ws, Ws_bf, n);
  }
  {
    int n = B_ * H_;
    init_state<<<(n + 255) / 256, 256, 0, stream>>>(hA, hB, h32, c32, n);
  }

  unsigned short* hrd = hA;
  unsigned short* hwr = hB;
  for (int t = 0; t < T_; ++t) {
    lstm_step<<<64, 256, 0, stream>>>(x_bf, Wi_bf, Ws_bf, bias, lens,
                                      hrd, hwr, h32, c32, out, t);
    unsigned short* tmp = hrd; hrd = hwr; hwr = tmp;
  }

  {
    int n = B_ * H_;
    copy_finals<<<(n + 255) / 256, 256, 0, stream>>>(h32, c32,
        out + (size_t)B_ * T_ * H_, n);
  }
}
